// CrossAttention_30056181138117
// MI455X (gfx1250) — matrix operand order repaired and checked by reading, not compiled, not run on hardware
//
#include <hip/hip_runtime.h>

// ---------------------------------------------------------------------------
// CrossAttention (seq_len==1) on MI455X / gfx1250.
// softmax over a length-1 axis == 1.0, so attn == V projection (Wq/Wk dead).
// Per branch: v = kv@Wv^T+bv ; xsum = v@Wo^T+bo+x ; x1 = LN(xsum) ;
//             h = relu(x1@W1^T+b1) ; ysum = h@W2^T+b2+x1 ; out = LN(ysum)
// GEMMs: v_wmma_f32_16x16x32_bf16, double-buffered LDS fed by
// global_load_async_to_lds_b128 (ASYNCcnt), residual fused in epilogue.
// ---------------------------------------------------------------------------

typedef __attribute__((ext_vector_type(16))) __bf16 bf16x16;
typedef __attribute__((ext_vector_type(8)))  __bf16 bf16x8;
typedef __attribute__((ext_vector_type(8)))  float  f32x8;

__device__ __forceinline__ __bf16 f2bf(float f) {
  unsigned int u = __builtin_bit_cast(unsigned int, f);
  unsigned int r = u + 0x7FFFu + ((u >> 16) & 1u);     // round-to-nearest-even
  unsigned short h = (unsigned short)(r >> 16);
  return __builtin_bit_cast(__bf16, h);
}

// ---------------------------------------------------------------------------
// f32 -> bf16 conversion, 8 elems/thread (weights + activations)
// ---------------------------------------------------------------------------
__global__ __launch_bounds__(256) void cvt8_f32_bf16(
    const float* __restrict__ in, __bf16* __restrict__ out, int n8) {
  int i = blockIdx.x * blockDim.x + threadIdx.x;
  if (i < n8) {
    const float4* p = reinterpret_cast<const float4*>(in + (size_t)i * 8);
    float4 a = p[0], b = p[1];
    bf16x8 v;
    v[0] = f2bf(a.x); v[1] = f2bf(a.y); v[2] = f2bf(a.z); v[3] = f2bf(a.w);
    v[4] = f2bf(b.x); v[5] = f2bf(b.y); v[6] = f2bf(b.z); v[7] = f2bf(b.w);
    *reinterpret_cast<bf16x8*>(out + (size_t)i * 8) = v;
  }
}

// ---------------------------------------------------------------------------
// WMMA GEMM:  C[M,N] = A[M,K] @ W[N,K]^T + bias[N] (+res) (opt. ReLU)
// A, W bf16 row-major. 256 threads (8 wave32), tile 128x128, BK=32.
// Double-buffered LDS filled by global_load_async_to_lds_b128.
// Wave tile 64(M) x 32(N) = 4x2 accumulators of 16x16.
// ---------------------------------------------------------------------------
constexpr int BM = 128, BN = 128, BK = 32;
constexpr int LDT = 40;   // padded LDS row stride (elems) = 80B: 16B-aligned,
                          // conflict-free for 16-lane strided b128 reads

template <bool RELU, bool RES, bool OUTBF>
__global__ __launch_bounds__(256) void gemm_bias(
    const __bf16* __restrict__ A, const __bf16* __restrict__ W,
    const float* __restrict__ bias, const float* res, void* Cv,
    int M, int N, int K)
{
  __shared__ __bf16 lA[2][BM * LDT];   // 2 x 10240 B
  __shared__ __bf16 lB[2][BN * LDT];   // 2 x 10240 B   (total 40 KB)

  const int tid   = threadIdx.x;
  const int lane  = tid & 31;
  const int wid   = tid >> 5;
  const int waveM = wid >> 2;        // 0..1
  const int waveN = wid & 3;         // 0..3
  const int m0 = blockIdx.x * BM;
  const int n0 = blockIdx.y * BN;

  // loader mapping: 2 threads per row, 16 contiguous k-elems (32B) each
  const int lrow = tid >> 1;          // 0..127
  const int lks  = (tid & 1) * 16;    // 0 or 16

  // ISA 7.12.2 fragment lane mapping
  const int l15 = lane & 15;
  const int hk  = (lane >> 4) * 8;    // A: upper lane-half holds K+8 chunk
  const int bk  = (lane >> 4) * 8;    // B: upper lane-half holds K+8 chunk

  f32x8 acc[4][2];
#pragma unroll
  for (int i = 0; i < 4; ++i)
#pragma unroll
    for (int j = 0; j < 2; ++j)
#pragma unroll
      for (int r = 0; r < 8; ++r) acc[i][j][r] = 0.0f;

  const __bf16* gA = A + (size_t)(m0 + lrow) * K + lks;
  const __bf16* gW = W + (size_t)(n0 + lrow) * K + lks;

  // issue async copies of one 128x32 A panel + 128x32 B panel into buffer b
  auto issue = [&](int b, int k0) {
    unsigned la = (unsigned)(size_t)&lA[b][lrow * LDT + lks];
    unsigned lb = (unsigned)(size_t)&lB[b][lrow * LDT + lks];
    unsigned long long ga = (unsigned long long)(gA + k0);
    unsigned long long gb = (unsigned long long)(gW + k0);
    asm volatile(
        "global_load_async_to_lds_b128 %0, %2, off\n\t"
        "global_load_async_to_lds_b128 %0, %2, off offset:16\n\t"
        "global_load_async_to_lds_b128 %1, %3, off\n\t"
        "global_load_async_to_lds_b128 %1, %3, off offset:16"
        :: "v"(la), "v"(lb), "v"(ga), "v"(gb) : "memory");
  };

  // one K-step of 32: 4 A frags, 2 B frags, 8 WMMAs
  auto compute = [&](int b) {
    bf16x16 af[4], bfg[2];
#pragma unroll
    for (int i = 0; i < 4; ++i) {
      const __bf16* base = &lA[b][(waveM * 64 + i * 16 + l15) * LDT + hk];
      bf16x8 c0 = *reinterpret_cast<const bf16x8*>(base);       // K = hk+0..7
      bf16x8 c1 = *reinterpret_cast<const bf16x8*>(base + 16);  // K = hk+16..23
      af[i] = __builtin_shufflevector(c0, c1, 0,1,2,3,4,5,6,7,8,9,10,11,12,13,14,15);
    }
#pragma unroll
    for (int j = 0; j < 2; ++j) {
      const __bf16* base = &lB[b][(waveN * 32 + j * 16 + l15) * LDT + bk];
      bf16x8 c0 = *reinterpret_cast<const bf16x8*>(base);       // K = bk+0..7
      bf16x8 c1 = *reinterpret_cast<const bf16x8*>(base + 16);  // K = bk+16..23
      bfg[j] = __builtin_shufflevector(c0, c1, 0,1,2,3,4,5,6,7,8,9,10,11,12,13,14,15);
    }
#pragma unroll
    for (int i = 0; i < 4; ++i)
#pragma unroll
      for (int j = 0; j < 2; ++j)
        acc[i][j] = __builtin_amdgcn_wmma_f32_16x16x32_bf16(
            false, af[i], false, bfg[j], (short)0, acc[i][j], false, false);
  };

  const int T = K / BK;                // 16 or 32 (always even)
  issue(0, 0);
  for (int kt = 0; kt < T; kt += 2) {
    issue(1, (kt + 1) * BK);
    asm volatile("s_wait_asynccnt 0x4" ::: "memory");   // buffer-0 panel resident
    __syncthreads();
    compute(0);
    __syncthreads();                                    // all waves done with buf0
    if (kt + 2 < T) {
      issue(0, (kt + 2) * BK);
      asm volatile("s_wait_asynccnt 0x4" ::: "memory"); // buffer-1 panel resident
    } else {
      asm volatile("s_wait_asynccnt 0x0" ::: "memory");
    }
    __syncthreads();
    compute(1);
    __syncthreads();
  }

  // ---- epilogue: bias (+residual, +relu), store f32 or bf16 ----
#pragma unroll
  for (int j = 0; j < 2; ++j) {
    const int n = n0 + waveN * 32 + j * 16 + l15;
    const float bv = bias[n];
#pragma unroll
    for (int i = 0; i < 4; ++i) {
      const int mbase = m0 + waveM * 64 + i * 16 + ((lane >> 4) << 3);
#pragma unroll
      for (int r = 0; r < 8; ++r) {
        const size_t idx = (size_t)(mbase + r) * N + n;
        float v = acc[i][j][r] + bv;
        if constexpr (RES) v += res[idx];
        if constexpr (RELU) v = v > 0.0f ? v : 0.0f;
        if constexpr (OUTBF) ((__bf16*)Cv)[idx] = f2bf(v);
        else                 ((float*)Cv)[idx]  = v;
      }
    }
  }
}

// ---------------------------------------------------------------------------
// LayerNorm over a 512-elem row (one wave32 per row, 16 elems/lane).
// outf[row*ldo+coff+c] = LN(xin[row])*g + be ; optional bf16 copy (outbf).
// In-place (outf == xin) is safe: whole row is read before writing.
// ---------------------------------------------------------------------------
template <bool BFOUT>
__global__ __launch_bounds__(256) void add_ln(
    const float* xin, const float* __restrict__ g, const float* __restrict__ be,
    float* outf, int ldo, int coff, __bf16* outbf)
{
  const int lane = threadIdx.x & 31;
  const int wid  = threadIdx.x >> 5;
  const int row  = blockIdx.x * 8 + wid;
  const int c0   = lane * 16;

  const float4* pa = reinterpret_cast<const float4*>(xin + (size_t)row * 512 + c0);
  float x[16];
#pragma unroll
  for (int c = 0; c < 4; ++c) {
    float4 a = pa[c];
    x[4 * c + 0] = a.x; x[4 * c + 1] = a.y; x[4 * c + 2] = a.z; x[4 * c + 3] = a.w;
  }
  float s = 0.0f, ss = 0.0f;
#pragma unroll
  for (int i = 0; i < 16; ++i) { s += x[i]; ss += x[i] * x[i]; }
#pragma unroll
  for (int m = 16; m >= 1; m >>= 1) {
    s  += __shfl_xor(s,  m, 32);
    ss += __shfl_xor(ss, m, 32);
  }
  const float mean = s * (1.0f / 512.0f);
  const float var  = ss * (1.0f / 512.0f) - mean * mean;
  const float rstd = rsqrtf(var + 1e-5f);

  const float4* pg  = reinterpret_cast<const float4*>(g  + c0);
  const float4* pbe = reinterpret_cast<const float4*>(be + c0);
  float4* po = reinterpret_cast<float4*>(outf + (size_t)row * ldo + coff + c0);
  float y[16];
#pragma unroll
  for (int c = 0; c < 4; ++c) {
    float4 gg = pg[c], bb = pbe[c], o;
    o.x = (x[4 * c + 0] - mean) * rstd * gg.x + bb.x;
    o.y = (x[4 * c + 1] - mean) * rstd * gg.y + bb.y;
    o.z = (x[4 * c + 2] - mean) * rstd * gg.z + bb.z;
    o.w = (x[4 * c + 3] - mean) * rstd * gg.w + bb.w;
    po[c] = o;
    y[4 * c + 0] = o.x; y[4 * c + 1] = o.y; y[4 * c + 2] = o.z; y[4 * c + 3] = o.w;
  }
  if constexpr (BFOUT) {
    __bf16* pb = outbf + (size_t)row * 512 + c0;
#pragma unroll
    for (int h = 0; h < 2; ++h) {
      bf16x8 v;
#pragma unroll
      for (int e = 0; e < 8; ++e) v[e] = f2bf(y[8 * h + e]);
      *reinterpret_cast<bf16x8*>(pb + 8 * h) = v;
    }
  }
}

// ---------------------------------------------------------------------------
// Host-side orchestration
// ---------------------------------------------------------------------------
extern "C" void kernel_launch(void* const* d_in, const int* in_sizes, int n_in,
                              void* d_out, int out_size, void* d_ws, size_t ws_size,
                              hipStream_t stream) {
  constexpr int E = 512, F = 1024, Mrows = 32768;

  const float* dna      = (const float*)d_in[0];
  const float* mol      = (const float*)d_in[1];
  const float* a1_in_w  = (const float*)d_in[2];
  const float* a1_in_b  = (const float*)d_in[3];
  const float* a1_out_w = (const float*)d_in[4];
  const float* a1_out_b = (const float*)d_in[5];
  const float* a2_in_w  = (const float*)d_in[6];
  const float* a2_in_b  = (const float*)d_in[7];
  const float* a2_out_w = (const float*)d_in[8];
  const float* a2_out_b = (const float*)d_in[9];
  const float* ln1_g = (const float*)d_in[10]; const float* ln1_b = (const float*)d_in[11];
  const float* ln2_g = (const float*)d_in[12]; const float* ln2_b = (const float*)d_in[13];
  const float* ln3_g = (const float*)d_in[14]; const float* ln3_b = (const float*)d_in[15];
  const float* ln4_g = (const float*)d_in[16]; const float* ln4_b = (const float*)d_in[17];
  const float* f1_w1 = (const float*)d_in[18]; const float* f1_b1 = (const float*)d_in[19];
  const float* f1_w2 = (const float*)d_in[20]; const float* f1_b2 = (const float*)d_in[21];
  const float* f2_w1 = (const float*)d_in[22]; const float* f2_b1 = (const float*)d_in[23];
  const float* f2_w2 = (const float*)d_in[24]; const float* f2_b2 = (const float*)d_in[25];

  // workspace carve-out
  size_t off = 0;
  auto alloc = [&](size_t bytes) -> void* {
    void* p = (char*)d_ws + off;
    off += (bytes + 255) & ~(size_t)255;
    return p;
  };
  __bf16* wv1 = (__bf16*)alloc((size_t)E * E * 2);
  __bf16* wo1 = (__bf16*)alloc((size_t)E * E * 2);
  __bf16* w11 = (__bf16*)alloc((size_t)F * E * 2);
  __bf16* w12 = (__bf16*)alloc((size_t)E * F * 2);
  __bf16* wv2 = (__bf16*)alloc((size_t)E * E * 2);
  __bf16* wo2 = (__bf16*)alloc((size_t)E * E * 2);
  __bf16* w21 = (__bf16*)alloc((size_t)F * E * 2);
  __bf16* w22 = (__bf16*)alloc((size_t)E * F * 2);
  __bf16* dnabf = (__bf16*)alloc((size_t)Mrows * E * 2);
  __bf16* molbf = (__bf16*)alloc((size_t)Mrows * E * 2);
  __bf16* vbf   = (__bf16*)alloc((size_t)Mrows * E * 2);  // V projection
  __bf16* x1bf  = (__bf16*)alloc((size_t)Mrows * E * 2);  // post-LN1 (bf16)
  __bf16* hbf   = (__bf16*)alloc((size_t)Mrows * F * 2);  // FFN hidden
  float*  xsum  = (float*)alloc((size_t)Mrows * E * 4);   // residual sums (f32)
  (void)ws_size; (void)n_in; (void)in_sizes; (void)out_size;

  auto cvt = [&](const float* src, __bf16* dst, size_t n) {
    int n8 = (int)(n / 8);
    cvt8_f32_bf16<<<(n8 + 255) / 256, 256, 0, stream>>>(src, dst, n8);
  };
  // Wq/Wk are dead: softmax over a length-1 axis == 1
  cvt(a1_in_w + 2 * E * E, wv1, (size_t)E * E);
  cvt(a1_out_w,            wo1, (size_t)E * E);
  cvt(f1_w1,               w11, (size_t)F * E);
  cvt(f1_w2,               w12, (size_t)E * F);
  cvt(a2_in_w + 2 * E * E, wv2, (size_t)E * E);
  cvt(a2_out_w,            wo2, (size_t)E * E);
  cvt(f2_w1,               w21, (size_t)F * E);
  cvt(f2_w2,               w22, (size_t)E * F);
  cvt(dna, dnabf, (size_t)Mrows * E);
  cvt(mol, molbf, (size_t)Mrows * E);

  const dim3 gE(Mrows / BM, E / BN);
  const dim3 gF(Mrows / BM, F / BN);
  const int  lnBlocks = Mrows / 8;

  auto run_branch = [&](const float* x_self, const __bf16* x_other_bf,
                        const __bf16* wv, const float* bv,
                        const __bf16* wo, const float* bo,
                        const __bf16* w1, const float* b1,
                        const __bf16* w2, const float* b2,
                        const float* g1, const float* be1,
                        const float* g3, const float* be3, int coff) {
    // v = x_other @ Wv^T + bv                       (bf16)
    gemm_bias<false, false, true ><<<gE, 256, 0, stream>>>(x_other_bf, wv, bv, nullptr, vbf, Mrows, E, E);
    // xsum = v @ Wo^T + bo + x_self                 (f32)
    gemm_bias<false, true,  false><<<gE, 256, 0, stream>>>(vbf, wo, bo, x_self, xsum, Mrows, E, E);
    // x1 = LN(xsum) -> xsum (in-place f32) + x1bf (bf16)
    add_ln<true ><<<lnBlocks, 256, 0, stream>>>(xsum, g1, be1, xsum, E, 0, x1bf);
    // h = relu(x1 @ W1^T + b1)                      (bf16)
    gemm_bias<true,  false, true ><<<gF, 256, 0, stream>>>(x1bf, w1, b1, nullptr, hbf, Mrows, F, E);
    // ysum = h @ W2^T + b2 + x1  (in-place over xsum)
    gemm_bias<false, true,  false><<<gE, 256, 0, stream>>>(hbf, w2, b2, xsum, xsum, Mrows, E, F);
    // out[:, coff:coff+E] = LN(ysum)
    add_ln<false><<<lnBlocks, 256, 0, stream>>>(xsum, g3, be3, (float*)d_out, 2 * E, coff, nullptr);
  };

  run_branch(dna, molbf, wv1, a1_in_b + 2 * E, wo1, a1_out_b,
             w11, f1_b1, w12, f1_b2, ln1_g, ln1_b, ln3_g, ln3_b, 0);
  run_branch(mol, dnabf, wv2, a2_in_b + 2 * E, wo2, a2_out_b,
             w21, f2_b1, w22, f2_b2, ln2_g, ln2_b, ln4_g, ln4_b, E);
}
